// PointTransformerCls_50268297232501
// MI455X (gfx1250) — matrix operand order repaired and checked by reading, not compiled, not run on hardware
//
#include <hip/hip_runtime.h>
#include <hip/hip_fp16.h>
#include <math.h>

// ---------------------------------------------------------------------------
// PointTransformer forward for gfx1250 (MI455X).
// All dense math goes through one WMMA GEMM kernel:
//   v_wmma_f32_16x16x32_f16, f32 accumulate, f32->f16 staged via LDS.
// Staging is two-phase (all global loads first, then cvt+LDS stores) so loads
// pipeline under a single s_wait_loadcnt; interior tiles use b128 loads and
// b64 LDS stores; edge tiles use branch-free clamped-address staging.
// ---------------------------------------------------------------------------

typedef _Float16 v16h __attribute__((ext_vector_type(16)));
typedef _Float16 h4   __attribute__((ext_vector_type(4)));
typedef float    v8f  __attribute__((ext_vector_type(8)));

union FragH { v16h v; _Float16 h[16]; };
union FragF { v8f  v; float    f[8];  };

#define GEMM_BM 64
#define GEMM_BN 64
#define GEMM_BK 32

// C[M,N] = act(scale * A[M,K] @ opB(B) + bias + R)
//   opB==0: B is [K,N] row-major (ldb); opB==1: B is [N,K] row-major (ldb) -> B^T
// 8 waves/block; each wave owns a 16x32 C slab (2 accumulators, A-fragment reuse).
__global__ __launch_bounds__(256)
void k_gemm(const float* __restrict__ A, int lda,
            const float* __restrict__ B, int ldb, int opB,
            const float* __restrict__ bias,
            const float* __restrict__ R, int ldr,
            float* __restrict__ C, int ldc,
            int M, int N, int K, float scale, int relu)
{
    __shared__ _Float16 lA[GEMM_BM * GEMM_BK];
    __shared__ _Float16 lB[GEMM_BN * GEMM_BK];   // stored transposed: [n][k]

    const int t    = threadIdx.x;
    const int lane = t & 31;
    const int wid  = t >> 5;
    const int wm   = wid >> 1;   // 0..3 (M tile)
    const int wn   = wid & 1;    // 0..1 (N tile)
    const int rowBase = blockIdx.y * GEMM_BM;
    const int colBase = blockIdx.x * GEMM_BN;

    const int hi   = (lane & 16) ? 1 : 0;         // lane-half
    const int mrow = (lane & 15) + wm * 16;       // A row within block tile
    const int n0   = wn * 32 + (lane & 15);       // first B column of this lane

    // uniform alignment predicates (allow b128 global loads on interior tiles)
    const bool a4 = ((((unsigned long)A) & 15) == 0) && ((lda & 3) == 0);
    const bool b4 = ((((unsigned long)B) & 15) == 0) && ((ldb & 3) == 0);

    FragF acc0, acc1;
#pragma unroll
    for (int j = 0; j < 8; ++j) { acc0.f[j] = 0.f; acc1.f[j] = 0.f; }

    // ---- guard-free staging for fully interior tiles (two-phase) ----
    auto stage_fast = [&](int k0, bool pf) {
        // phase 1: all global loads into registers
        float va[2][4], vb[2][4];
#pragma unroll
        for (int i = 0; i < 2; ++i) {
            int q = i * 256 + t;                  // 0..511 quads
            int r = q >> 3, kc = (q & 7) << 2;    // 4 consecutive K per thread
            const float* src = A + (long)(rowBase + r) * lda + (k0 + kc);
            if (pf && i == 0) __builtin_prefetch(src + GEMM_BK, 0, 1);
            if (a4) {
                float4 v4 = *(const float4*)src;
                va[i][0] = v4.x; va[i][1] = v4.y; va[i][2] = v4.z; va[i][3] = v4.w;
            } else {
                va[i][0] = src[0]; va[i][1] = src[1]; va[i][2] = src[2]; va[i][3] = src[3];
            }
        }
        if (opB == 0) {
#pragma unroll
            for (int i = 0; i < 2; ++i) {
                int q = i * 256 + t;
                int n = q & 63, kc = (q >> 6) << 2;    // fixed col, 4 K rows
                const float* src = B + (long)(k0 + kc) * ldb + (colBase + n);
                if (pf && i == 0) __builtin_prefetch(src + (long)GEMM_BK * ldb, 0, 1);
                vb[i][0] = src[0];
                vb[i][1] = src[(long)ldb];
                vb[i][2] = src[(long)ldb * 2];
                vb[i][3] = src[(long)ldb * 3];
            }
        } else {
#pragma unroll
            for (int i = 0; i < 2; ++i) {
                int q = i * 256 + t;
                int n = q >> 3, kc = (q & 7) << 2;     // 4 consecutive K per thread
                const float* src = B + (long)(colBase + n) * ldb + (k0 + kc);
                if (pf && i == 0) __builtin_prefetch(src + GEMM_BK, 0, 1);
                if (b4) {
                    float4 v4 = *(const float4*)src;
                    vb[i][0] = v4.x; vb[i][1] = v4.y; vb[i][2] = v4.z; vb[i][3] = v4.w;
                } else {
                    vb[i][0] = src[0]; vb[i][1] = src[1]; vb[i][2] = src[2]; vb[i][3] = src[3];
                }
            }
        }
        // phase 2: cvt + vector LDS stores
#pragma unroll
        for (int i = 0; i < 2; ++i) {
            int q = i * 256 + t;
            int r = q >> 3, kc = (q & 7) << 2;
            h4 hv;
            hv[0] = (_Float16)va[i][0]; hv[1] = (_Float16)va[i][1];
            hv[2] = (_Float16)va[i][2]; hv[3] = (_Float16)va[i][3];
            *(h4*)(&lA[r * GEMM_BK + kc]) = hv;
        }
        if (opB == 0) {
#pragma unroll
            for (int i = 0; i < 2; ++i) {
                int q = i * 256 + t;
                int n = q & 63, kc = (q >> 6) << 2;
                h4 hv;
                hv[0] = (_Float16)vb[i][0]; hv[1] = (_Float16)vb[i][1];
                hv[2] = (_Float16)vb[i][2]; hv[3] = (_Float16)vb[i][3];
                *(h4*)(&lB[n * GEMM_BK + kc]) = hv;
            }
        } else {
#pragma unroll
            for (int i = 0; i < 2; ++i) {
                int q = i * 256 + t;
                int n = q >> 3, kc = (q & 7) << 2;
                h4 hv;
                hv[0] = (_Float16)vb[i][0]; hv[1] = (_Float16)vb[i][1];
                hv[2] = (_Float16)vb[i][2]; hv[3] = (_Float16)vb[i][3];
                *(h4*)(&lB[n * GEMM_BK + kc]) = hv;
            }
        }
    };

    // ---- branch-free guarded staging (clamped address, two-phase) ----
    auto stage_guard = [&](int k0) {
        float va[8], vb[8];
        bool  oa[8], ob[8];
#pragma unroll
        for (int i = 0; i < 8; ++i) {
            int lin = i * 256 + t;
            int r = lin >> 5, kk = lin & 31;
            int gr = rowBase + r, gk = k0 + kk;
            oa[i] = (gr < M) && (gk < K);
            va[i] = A[oa[i] ? ((long)gr * lda + gk) : 0];
        }
        if (opB == 0) {
#pragma unroll
            for (int i = 0; i < 8; ++i) {
                int lin = i * 256 + t;
                int kk = lin >> 6, n = lin & 63;
                int gk = k0 + kk, gn = colBase + n;
                ob[i] = (gk < K) && (gn < N);
                vb[i] = B[ob[i] ? ((long)gk * ldb + gn) : 0];
            }
        } else {
#pragma unroll
            for (int i = 0; i < 8; ++i) {
                int lin = i * 256 + t;
                int n = lin >> 5, kk = lin & 31;
                int gk = k0 + kk, gn = colBase + n;
                ob[i] = (gk < K) && (gn < N);
                vb[i] = B[ob[i] ? ((long)gn * ldb + gk) : 0];
            }
        }
#pragma unroll
        for (int i = 0; i < 8; ++i) {
            int lin = i * 256 + t;
            int r = lin >> 5, kk = lin & 31;
            lA[r * GEMM_BK + kk] = (_Float16)(oa[i] ? va[i] : 0.f);
        }
        if (opB == 0) {
#pragma unroll
            for (int i = 0; i < 8; ++i) {
                int lin = i * 256 + t;
                int kk = lin >> 6, n = lin & 63;
                lB[n * GEMM_BK + kk] = (_Float16)(ob[i] ? vb[i] : 0.f);
            }
        } else {
#pragma unroll
            for (int i = 0; i < 8; ++i) {
                int lin = i * 256 + t;
                int n = lin >> 5, kk = lin & 31;
                lB[n * GEMM_BK + kk] = (_Float16)(ob[i] ? vb[i] : 0.f);
            }
        }
    };

    // ---- fragments + WMMA (A 16x32 f16 layout; B mirrored per ISA) ----
    auto compute = [&]() {
        FragH fa, fb0, fb1;
#pragma unroll
        for (int v = 0; v < 8; ++v) {
            int ka = hi * 16 + (v << 1);
            fa.h[2 * v]     = lA[mrow * GEMM_BK + ka];
            fa.h[2 * v + 1] = lA[mrow * GEMM_BK + ka + 1];
            int kb = hi * 16 + (v << 1);
            fb0.h[2 * v]     = lB[n0 * GEMM_BK + kb];
            fb0.h[2 * v + 1] = lB[n0 * GEMM_BK + kb + 1];
            fb1.h[2 * v]     = lB[(n0 + 16) * GEMM_BK + kb];
            fb1.h[2 * v + 1] = lB[(n0 + 16) * GEMM_BK + kb + 1];
        }
        acc0.v = __builtin_amdgcn_wmma_f32_16x16x32_f16(false, fa.v, false, fb0.v,
                                                        (short)0, acc0.v, false, false);
        acc1.v = __builtin_amdgcn_wmma_f32_16x16x32_f16(false, fa.v, false, fb1.v,
                                                        (short)0, acc1.v, false, false);
    };

    auto epilogue = [&](bool guard) {
        const int rbase = rowBase + wm * 16 + hi * 8;
        const int c0 = colBase + wn * 32 + (lane & 15);
#pragma unroll
        for (int j = 0; j < 8; ++j) {
            int rr = rbase + j;
            if (guard && rr >= M) continue;
#pragma unroll
            for (int s2 = 0; s2 < 2; ++s2) {
                int cc = c0 + s2 * 16;
                if (guard && cc >= N) continue;
                float v = (s2 ? acc1.f[j] : acc0.f[j]) * scale;
                if (bias) v += bias[cc];
                if (R)    v += R[(long)rr * ldr + cc];
                if (relu) v = fmaxf(v, 0.f);
                C[(long)rr * ldc + cc] = v;
            }
        }
    };

    const bool fullMN = (rowBase + GEMM_BM <= M) && (colBase + GEMM_BN <= N);
    if (fullMN) {
        const int kfull = K & ~(GEMM_BK - 1);
        for (int k0 = 0; k0 < kfull; k0 += GEMM_BK) {
            stage_fast(k0, k0 + GEMM_BK < kfull);
            __syncthreads();
            compute();
            __syncthreads();
        }
        if (kfull < K) {
            stage_guard(kfull);
            __syncthreads();
            compute();
            __syncthreads();
        }
        epilogue(false);
    } else {
        for (int k0 = 0; k0 < K; k0 += GEMM_BK) {
            stage_guard(k0);
            __syncthreads();
            compute();
            __syncthreads();
        }
        epilogue(true);
    }
}

// Row softmax, one wave (32 lanes) per row.
__global__ void k_softmax(float* __restrict__ X, int rows, int L)
{
    int wid = threadIdx.x >> 5, lane = threadIdx.x & 31;
    int row = blockIdx.x * (blockDim.x >> 5) + wid;
    if (row >= rows) return;
    float* x = X + (long)row * L;
    float m = -3.4e38f;
    for (int c = lane; c < L; c += 32) m = fmaxf(m, x[c]);
#pragma unroll
    for (int o = 16; o; o >>= 1) m = fmaxf(m, __shfl_xor(m, o, 32));
    float s = 0.f;
    for (int c = lane; c < L; c += 32) { float e = expf(x[c] - m); x[c] = e; s += e; }
#pragma unroll
    for (int o = 16; o; o >>= 1) s += __shfl_xor(s, o, 32);
    float inv = 1.f / s;
    for (int c = lane; c < L; c += 32) x[c] *= inv;
}

// LayerNorm over last dim, one wave per row.
__global__ void k_layernorm(const float* __restrict__ X, const float* __restrict__ g,
                            const float* __restrict__ bet, float* __restrict__ Y,
                            int rows, int dim)
{
    int wid = threadIdx.x >> 5, lane = threadIdx.x & 31;
    int row = blockIdx.x * (blockDim.x >> 5) + wid;
    if (row >= rows) return;
    const float* x = X + (long)row * dim;
    float s = 0.f;
    for (int c = lane; c < dim; c += 32) s += x[c];
#pragma unroll
    for (int o = 16; o; o >>= 1) s += __shfl_xor(s, o, 32);
    float mu = s / dim;
    float v = 0.f;
    for (int c = lane; c < dim; c += 32) { float d = x[c] - mu; v += d * d; }
#pragma unroll
    for (int o = 16; o; o >>= 1) v += __shfl_xor(v, o, 32);
    float inv = 1.f / sqrtf(v / dim + 1e-6f);
    float* y = Y + (long)row * dim;
    for (int c = lane; c < dim; c += 32) y[c] = (x[c] - mu) * inv * g[c] + bet[c];
}

// Iterative top-k (matches lax.top_k ordering: descending, ties -> lower index).
__global__ void k_topk(const float* __restrict__ score, int* __restrict__ out, int k)
{
    int b = blockIdx.x, t = threadIdx.x;
    __shared__ float s[1024];
    __shared__ float rv[256];
    __shared__ int   ri[256];
    for (int n = t; n < 1024; n += 256) s[n] = score[b * 1024 + n];
    for (int i = 0; i < k; ++i) {
        __syncthreads();
        float best = -3.4e38f; int bi = 0;
        for (int n = t; n < 1024; n += 256)
            if (s[n] > best) { best = s[n]; bi = n; }
        rv[t] = best; ri[t] = bi;
        __syncthreads();
        for (int w = 128; w; w >>= 1) {
            if (t < w) {
                if (rv[t + w] > rv[t] || (rv[t + w] == rv[t] && ri[t + w] < ri[t])) {
                    rv[t] = rv[t + w]; ri[t] = ri[t + w];
                }
            }
            __syncthreads();
        }
        if (t == 0) { out[b * k + i] = ri[0]; s[ri[0]] = -3.4e38f; }
    }
}

// Farthest point sampling; one block per batch, dist kept in registers (4/lane).
__global__ void k_fps(const float* __restrict__ pts, int* __restrict__ idxs, int npoint)
{
    int b = blockIdx.x, t = threadIdx.x;
    __shared__ float rv[256];
    __shared__ int   ri[256];
    __shared__ int   sfar;
    float dist[4];
#pragma unroll
    for (int j = 0; j < 4; ++j) dist[j] = 1e10f;
    int far = 0;
    for (int i = 0; i < npoint; ++i) {
        if (t == 0) idxs[b * npoint + i] = far;
        const float* c = pts + ((long)b * 1024 + far) * 6;
        float cx = c[0], cy = c[1], cz = c[2];
        float best = -1.f; int bi = 0;
#pragma unroll
        for (int j = 0; j < 4; ++j) {
            int n = t * 4 + j;
            const float* p = pts + ((long)b * 1024 + n) * 6;
            float dx = p[0] - cx, dy = p[1] - cy, dz = p[2] - cz;
            float d = dx * dx + dy * dy + dz * dz;
            dist[j] = fminf(dist[j], d);
            if (dist[j] > best) { best = dist[j]; bi = n; }
        }
        rv[t] = best; ri[t] = bi;
        __syncthreads();
        for (int w = 128; w; w >>= 1) {
            if (t < w) {
                if (rv[t + w] > rv[t] || (rv[t + w] == rv[t] && ri[t + w] < ri[t])) {
                    rv[t] = rv[t + w]; ri[t] = ri[t + w];
                }
            }
            __syncthreads();
        }
        if (t == 0) sfar = ri[0];
        __syncthreads();
        far = sfar;
    }
}

// Ball query: first `ns` in-radius indices in ascending order, padded with first.
__global__ void k_ballquery(const float* __restrict__ pts, const int* __restrict__ seed,
                            int S, int ns, float r2, int* __restrict__ gidx)
{
    int id = blockIdx.x * blockDim.x + threadIdx.x;
    if (id >= 8 * S) return;
    int b = id / S, s = id - b * S;
    int sd = seed[b * S + s];
    const float* c = pts + ((long)b * 1024 + sd) * 6;
    float cx = c[0], cy = c[1], cz = c[2];
    int* out = gidx + (long)(b * S + s) * ns;
    int cnt = 0, first = -1;
    for (int n = 0; n < 1024 && cnt < ns; ++n) {
        const float* p = pts + ((long)b * 1024 + n) * 6;
        float dx = p[0] - cx, dy = p[1] - cy, dz = p[2] - cz;
        float d2 = dx * dx + dy * dy + dz * dz;
        if (d2 <= r2) { if (first < 0) first = n; out[cnt++] = n; }
    }
    if (first < 0) first = 0;
    for (int j = cnt; j < ns; ++j) out[j] = first;
}

// Build grouped features [S*ns, 515] = concat(feats[gidx], xyz[gidx]-new_xyz) for batch b.
__global__ void k_gather(const float* __restrict__ pts, const float* __restrict__ feats,
                         const int* __restrict__ gidx, const int* __restrict__ seed,
                         int b, int S, int ns, float* __restrict__ out)
{
    long total = (long)S * ns * 515;
    for (long id = blockIdx.x * (long)blockDim.x + threadIdx.x; id < total;
         id += (long)gridDim.x * blockDim.x) {
        long r = id / 515; int c = (int)(id - r * 515);
        int s = (int)(r / ns), j = (int)(r - (long)s * ns);
        int g = gidx[((long)b * S + s) * ns + j];
        float v;
        if (c < 512) v = feats[((long)b * 1024 + g) * 512 + c];
        else {
            int sd = seed[b * S + s];
            int cc = c - 512;
            v = pts[((long)b * 1024 + g) * 6 + cc] - pts[((long)b * 1024 + sd) * 6 + cc];
        }
        out[id] = v;
    }
}

// Max over neighbors; writes into sa_out column slice [coff, coff+C).
__global__ void k_maxpool(const float* __restrict__ in, int S, int ns, int C,
                          float* __restrict__ out, int ldo, int coff)
{
    int id = blockIdx.x * blockDim.x + threadIdx.x;
    if (id >= S * C) return;
    int s = id / C, c = id - s * C;
    float m = -3.4e38f;
    for (int j = 0; j < ns; ++j) m = fmaxf(m, in[((long)s * ns + j) * C + c]);
    out[(long)s * ldo + coff + c] = m;
}

// sortnet output: [agg(505) | score[topk](1) | points[topk](6)] -> local_features row.
__global__ void k_sncat(const float* __restrict__ agg, const float* __restrict__ score,
                        const float* __restrict__ pts, const int* __restrict__ topk,
                        int m, float* __restrict__ lf)
{
    int id = blockIdx.x * blockDim.x + threadIdx.x;
    if (id >= 8 * 64 * 512) return;
    int b = id / (64 * 512);
    int r = (id / 512) % 64;
    int c = id % 512;
    int tk = topk[b * 64 + r];
    float v;
    if (c < 505)       v = agg[((long)b * 64 + r) * 505 + c];
    else if (c == 505) v = score[b * 1024 + tk];
    else               v = pts[((long)b * 1024 + tk) * 6 + (c - 506)];
    lf[((long)b * 256 + m * 64 + r) * 512 + c] = v;
}

// ---------------------------------------------------------------------------
// Host-side orchestration
// ---------------------------------------------------------------------------

struct Lin { const float* W; const float* b; };
struct Mha { const float *ln_b, *ln_g, *w_k, *w_o, *w_q, *w_v; };
struct SNet { Lin fc[3]; Lin agg[3]; Lin sa[3][3]; };

static void read_lin(void* const* din, int& i, Lin* L, int n) {
    for (int j = 0; j < n; ++j) {
        L[j].W = (const float*)din[i++];
        L[j].b = (const float*)din[i++];
    }
}
static void read_mha(void* const* din, int& i, Mha& m) {
    m.ln_b = (const float*)din[i++]; m.ln_g = (const float*)din[i++];
    m.w_k  = (const float*)din[i++]; m.w_o  = (const float*)din[i++];
    m.w_q  = (const float*)din[i++]; m.w_v  = (const float*)din[i++];
}

static inline void gemm(hipStream_t st, const float* A, int lda,
                        const float* B, int ldb, int opB,
                        const float* bias, const float* R, int ldr,
                        float* C, int ldc, int M, int N, int K,
                        float scale, int relu)
{
    dim3 g((N + GEMM_BN - 1) / GEMM_BN, (M + GEMM_BM - 1) / GEMM_BM);
    k_gemm<<<g, 256, 0, st>>>(A, lda, B, ldb, opB, bias, R, ldr, C, ldc, M, N, K, scale, relu);
}

// Streaming per-(batch,head) MHA: out = LN(softmax(QK^T/sqrt(dh)) V @ w_o + qin)
static void run_mha(hipStream_t st, const Mha& p, const float* qin, int Lq,
                    const float* kvin, int Lk, float* out,
                    float* qh, float* kh, float* vh, float* attn,
                    float* ofull, float* pre)
{
    const float sc = 0.08838834764831845f;  // 1/sqrt(128)
    for (int b = 0; b < 8; ++b) {
        const float* Aq = qin  + (long)b * Lq * 512;
        const float* Ak = kvin + (long)b * Lk * 512;
        for (int h = 0; h < 4; ++h) {
            gemm(st, Aq, 512, p.w_q + h * 128, 512, 0, nullptr, nullptr, 0, qh, 128, Lq, 128, 512, 1.f, 0);
            gemm(st, Ak, 512, p.w_k + h * 128, 512, 0, nullptr, nullptr, 0, kh, 128, Lk, 128, 512, 1.f, 0);
            gemm(st, Ak, 512, p.w_v + h * 128, 512, 0, nullptr, nullptr, 0, vh, 128, Lk, 128, 512, 1.f, 0);
            gemm(st, qh, 128, kh, 128, 1, nullptr, nullptr, 0, attn, Lk, Lq, Lk, 128, sc, 0);
            k_softmax<<<dim3((Lq + 7) / 8), 256, 0, st>>>(attn, Lq, Lk);
            gemm(st, attn, Lk, vh, 128, 0, nullptr, nullptr, 0,
                 ofull + (long)b * Lq * 512 + h * 128, 512, Lq, 128, Lk, 1.f, 0);
        }
    }
    gemm(st, ofull, 512, p.w_o, 512, 0, nullptr, qin, 512, pre, 512, 8 * Lq, 512, 512, 1.f, 0);
    k_layernorm<<<dim3((8 * Lq + 7) / 8), 256, 0, st>>>(pre, p.ln_g, p.ln_b, out, 8 * Lq, 512);
}

// Multi-scale set abstraction -> sa_out [8, S, 320]
static void run_sa(hipStream_t st, const Lin (*sa)[3], const float* pts,
                   const float* feats, const int* seed, int S, float* sa_out,
                   int* gidx, float* gfeat, float* ping, float* pong)
{
    static const float R2[3] = {0.01f, 0.04f, 0.16f};          // radii^2
    static const int   NS[3] = {16, 32, 128};
    static const int   CH[3][3] = {{32, 32, 64}, {64, 64, 128}, {64, 96, 128}};
    static const int   COFF[3] = {0, 64, 192};
    for (int br = 0; br < 3; ++br) {
        int ns = NS[br];
        k_ballquery<<<dim3((8 * S + 255) / 256), 256, 0, st>>>(pts, seed, S, ns, R2[br], gidx);
        const int c0 = CH[br][0], c1 = CH[br][1], c2 = CH[br][2];
        for (int b = 0; b < 8; ++b) {
            long gtotal = (long)S * ns * 515;
            k_gather<<<dim3((unsigned)((gtotal + 255) / 256)), 256, 0, st>>>(
                pts, feats, gidx, seed, b, S, ns, gfeat);
            int rows = S * ns;
            gemm(st, gfeat, 515, sa[br][0].W, c0, 0, sa[br][0].b, nullptr, 0, ping, c0, rows, c0, 515, 1.f, 1);
            gemm(st, ping,  c0,  sa[br][1].W, c1, 0, sa[br][1].b, nullptr, 0, pong, c1, rows, c1, c0,  1.f, 1);
            gemm(st, pong,  c1,  sa[br][2].W, c2, 0, sa[br][2].b, nullptr, 0, ping, c2, rows, c2, c1,  1.f, 1);
            k_maxpool<<<dim3((S * c2 + 255) / 256), 256, 0, st>>>(
                ping, S, ns, c2, sa_out + (long)b * S * 320, 320, COFF[br]);
        }
    }
}

extern "C" void kernel_launch(void* const* d_in, const int* in_sizes, int n_in,
                              void* d_out, int out_size, void* d_ws, size_t ws_size,
                              hipStream_t stream)
{
    (void)in_sizes; (void)n_in; (void)out_size; (void)ws_size;

    // ---- parameter mapping (jax pytree leaf order: dict keys sorted) ----
    int i = 0;
    Lin cls[4];  read_lin(d_in, i, cls, 4);     // fc_cls
    Lin red[3];  read_lin(d_in, i, red, 3);     // fc_reduce
    Mha gatt;    read_mha(d_in, i, gatt);       // gfg_att
    Lin gfc[3];  read_lin(d_in, i, gfc, 3);     // gfg_fc
    Lin gagg[3]; read_lin(d_in, i, gagg, 3);    // gfg_fc_agg
    Lin gsa[3][3];
    for (int br = 0; br < 3; ++br) read_lin(d_in, i, gsa[br], 3);  // gfg_sa
    Mha latt;    read_mha(d_in, i, latt);       // lfg_att
    Lin lfc[3];  read_lin(d_in, i, lfc, 3);     // lfg_fc
    Mha lgatt;   read_mha(d_in, i, lgatt);      // lg_att
    SNet sn[4];                                  // sortnets: fc, fc_agg, sa
    for (int m = 0; m < 4; ++m) {
        read_lin(d_in, i, sn[m].fc, 3);
        read_lin(d_in, i, sn[m].agg, 3);
        for (int br = 0; br < 3; ++br) read_lin(d_in, i, sn[m].sa[br], 3);
    }
    const float* pts = (const float*)d_in[i++]; // points [8,1024,6]

    // ---- workspace arena (floats), ~98 MB ----
    float* w = (float*)d_ws;
    size_t off = 0;
    auto take = [&](size_t n) { float* p = w + off; off += n; return p; };
    float* x     = take(4194304);  // lfg features [8192,512]
    float* g     = take(4194304);  // gfg features [8192,512]
    float* bigA  = take(4218880);  // MHA O-buffer / SA gather (with bigB)
    float* bigB  = take(4218880);  // MHA pre-LN / SA gather tail
    float* ping  = take(2097152);  // generic [<=16384, <=256]
    float* pong  = take(2097152);
    float* qh    = take(131072);
    float* kh    = take(131072);
    float* vh    = take(131072);
    float* attn  = take(1048576);
    float* saout = take(327680);   // [8, <=128, 320]
    float* agg0  = take(524288);
    float* agg1  = take(524288);
    float* lf    = take(1048576);  // local features [8,256,512]
    float* gf    = take(524288);   // global features [8,128,512]
    float* scoreb= take(8192);     // sortnet scores [8,1024]
    float* xr    = take(131072);   // [8,256,64]
    float* cls0  = take(8192);
    float* cls1  = take(2048);
    float* cls2  = take(512);
    int* topk = (int*)take(512);   // [8,64]
    int* fpsi = (int*)take(1024);  // [8,128]
    int* gidx = (int*)take(131072);// [8,128,128]

    hipStream_t st = stream;

    // ================= Local Feature Generation =================
    gemm(st, pts, 6,    lfc[0].W, 64,  0, lfc[0].b, nullptr, 0, bigA, 64,  8192, 64,  6,   1.f, 1);
    gemm(st, bigA, 64,  lfc[1].W, 256, 0, lfc[1].b, nullptr, 0, bigB, 256, 8192, 256, 64,  1.f, 1);
    gemm(st, bigB, 256, lfc[2].W, 512, 0, lfc[2].b, nullptr, 0, x,    512, 8192, 512, 256, 1.f, 0);
    run_mha(st, latt, x, 1024, x, 1024, x, qh, kh, vh, attn, bigA, bigB);

    // sortnets -> local_features [8, 4*64, 512]
    for (int m = 0; m < 4; ++m) {
        gemm(st, x,    512, sn[m].fc[0].W, 256, 0, sn[m].fc[0].b, nullptr, 0, ping,   256, 8192, 256, 512, 1.f, 1);
        gemm(st, ping, 256, sn[m].fc[1].W, 64,  0, sn[m].fc[1].b, nullptr, 0, pong,   64,  8192, 64,  256, 1.f, 1);
        gemm(st, pong, 64,  sn[m].fc[2].W, 1,   0, sn[m].fc[2].b, nullptr, 0, scoreb, 1,   8192, 1,   64,  1.f, 0);
        k_topk<<<8, 256, 0, st>>>(scoreb, topk, 64);
        run_sa(st, sn[m].sa, pts, x, topk, 64, saout, gidx, bigA, ping, pong);
        gemm(st, saout, 320, sn[m].agg[0].W, 256, 0, sn[m].agg[0].b, nullptr, 0, agg0, 256, 512, 256, 320, 1.f, 1);
        gemm(st, agg0,  256, sn[m].agg[1].W, 256, 0, sn[m].agg[1].b, nullptr, 0, agg1, 256, 512, 256, 256, 1.f, 1);
        gemm(st, agg1,  256, sn[m].agg[2].W, 505, 0, sn[m].agg[2].b, nullptr, 0, agg0, 505, 512, 505, 256, 1.f, 0);
        k_sncat<<<dim3((8 * 64 * 512 + 255) / 256), 256, 0, st>>>(agg0, scoreb, pts, topk, m, lf);
    }

    // ================= Global Feature Generation =================
    gemm(st, pts, 6,    gfc[0].W, 64,  0, gfc[0].b, nullptr, 0, bigA, 64,  8192, 64,  6,   1.f, 1);
    gemm(st, bigA, 64,  gfc[1].W, 256, 0, gfc[1].b, nullptr, 0, bigB, 256, 8192, 256, 64,  1.f, 1);
    gemm(st, bigB, 256, gfc[2].W, 512, 0, gfc[2].b, nullptr, 0, g,    512, 8192, 512, 256, 1.f, 0);
    run_mha(st, gatt, g, 1024, g, 1024, g, qh, kh, vh, attn, bigA, bigB);
    k_fps<<<8, 256, 0, st>>>(pts, fpsi, 128);
    run_sa(st, gsa, pts, g, fpsi, 128, saout, gidx, bigA, ping, pong);
    gemm(st, saout, 320, gagg[0].W, 256, 0, gagg[0].b, nullptr, 0, agg0, 256, 1024, 256, 320, 1.f, 1);
    gemm(st, agg0,  256, gagg[1].W, 256, 0, gagg[1].b, nullptr, 0, agg1, 256, 1024, 256, 256, 1.f, 1);
    gemm(st, agg1,  256, gagg[2].W, 512, 0, gagg[2].b, nullptr, 0, gf,   512, 1024, 512, 256, 1.f, 0);

    // ================= Local-global fusion + classifier =================
    run_mha(st, lgatt, lf, 256, gf, 128, lf, qh, kh, vh, attn, bigA, bigB);
    gemm(st, lf,   512, red[0].W, 256, 0, red[0].b, nullptr, 0, agg0, 256, 2048, 256, 512, 1.f, 1);
    gemm(st, agg0, 256, red[1].W, 256, 0, red[1].b, nullptr, 0, agg1, 256, 2048, 256, 256, 1.f, 1);
    gemm(st, agg1, 256, red[2].W, 64,  0, red[2].b, nullptr, 0, xr,   64,  2048, 64,  256, 1.f, 0);

    float* out = (float*)d_out;
    gemm(st, xr,   16384, cls[0].W, 1024, 0, cls[0].b, nullptr, 0, cls0, 1024, 8, 1024, 16384, 1.f, 1);
    gemm(st, cls0, 1024,  cls[1].W, 256,  0, cls[1].b, nullptr, 0, cls1, 256,  8, 256,  1024,  1.f, 1);
    gemm(st, cls1, 256,   cls[2].W, 64,   0, cls[2].b, nullptr, 0, cls2, 64,   8, 64,   256,   1.f, 1);
    gemm(st, cls2, 64,    cls[3].W, 40,   0, cls[3].b, nullptr, 0, out,  40,   8, 40,   64,    1.f, 0);
}
